// SelfAttention_3298534883834
// MI455X (gfx1250) — matrix operand order repaired and checked by reading, not compiled, not run on hardware
//
#include <hip/hip_runtime.h>
#include <math.h>

typedef __attribute__((ext_vector_type(16))) _Float16 v16h;
typedef __attribute__((ext_vector_type(8)))  _Float16 v8h;
typedef __attribute__((ext_vector_type(4)))  _Float16 v4h;
typedef __attribute__((ext_vector_type(2)))  _Float16 v2h;
typedef __attribute__((ext_vector_type(8)))  float    v8f;

namespace {
constexpr int kNSeq  = 16;
constexpr int kL     = 1024;
constexpr int kH     = 16;
constexpr int kKD    = 64;
constexpr int kQKV   = 3 * kKD;        // 192
constexpr int kRow   = kH * kQKV;      // 3072 floats per qkv row
constexpr int kORow  = kH * kKD;       // 1024 floats per out row
constexpr int kChunk = 64;             // KV rows per iteration
constexpr int kIters = kL / kChunk;    // 16
constexpr int kWaves = 4;              // waves per block (4 Q-blocks of same (s,h))
constexpr int kStr   = 72;             // LDS row stride in halves (144 B, 16B-aligned rows)
}

// wave-private LDS producer->consumer ordering (CDNA5 split counters)
#define WAVE_LDS_SYNC()                                        \
  do {                                                         \
    __builtin_amdgcn_wave_barrier();                           \
    asm volatile("s_wait_dscnt 0x0" ::: "memory");             \
  } while (0)

// LDS halves p[0..7], p[16..23] (16B aligned) -> v16h B fragment
__device__ __forceinline__ v16h ld_frag16(const _Float16* p) {
  const v8h a = *reinterpret_cast<const v8h*>(p);
  const v8h b = *reinterpret_cast<const v8h*>(p + 16);
  return __builtin_shufflevector(a, b, 0, 1, 2, 3, 4, 5, 6, 7,
                                 8, 9, 10, 11, 12, 13, 14, 15);
}

// A-fragment from an LDS row: elems 0..7 at p[kb..kb+7], 8..15 at p[kb+16..kb+23]
__device__ __forceinline__ v16h ld_fragA(const _Float16* p, int kb) {
  const v8h a = *reinterpret_cast<const v8h*>(p + kb);
  const v8h b = *reinterpret_cast<const v8h*>(p + kb + 16);
  return __builtin_shufflevector(a, b, 0, 1, 2, 3, 4, 5, 6, 7,
                                 8, 9, 10, 11, 12, 13, 14, 15);
}

// 16 consecutive global floats -> 16 LDS halves (4x v4h stores, 8B aligned)
__device__ __forceinline__ void cvt16_store(const float* __restrict__ src,
                                            _Float16* dst) {
  const float4 a = *reinterpret_cast<const float4*>(src + 0);
  const float4 b = *reinterpret_cast<const float4*>(src + 4);
  const float4 c = *reinterpret_cast<const float4*>(src + 8);
  const float4 d = *reinterpret_cast<const float4*>(src + 12);
  v4h t0; t0[0] = (_Float16)a.x; t0[1] = (_Float16)a.y; t0[2] = (_Float16)a.z; t0[3] = (_Float16)a.w;
  v4h t1; t1[0] = (_Float16)b.x; t1[1] = (_Float16)b.y; t1[2] = (_Float16)b.z; t1[3] = (_Float16)b.w;
  v4h t2; t2[0] = (_Float16)c.x; t2[1] = (_Float16)c.y; t2[2] = (_Float16)c.z; t2[3] = (_Float16)c.w;
  v4h t3; t3[0] = (_Float16)d.x; t3[1] = (_Float16)d.y; t3[2] = (_Float16)d.z; t3[3] = (_Float16)d.w;
  *reinterpret_cast<v4h*>(dst + 0)  = t0;
  *reinterpret_cast<v4h*>(dst + 4)  = t1;
  *reinterpret_cast<v4h*>(dst + 8)  = t2;
  *reinterpret_cast<v4h*>(dst + 12) = t3;
}

// Load 8 consecutive floats, scale, place at v[o..o+7] (Q A-fragment helper)
__device__ __forceinline__ void fill8_scaled(const float* __restrict__ p, float sc,
                                             v16h& v, int o) {
  const float4 a = *reinterpret_cast<const float4*>(p + 0);
  const float4 b = *reinterpret_cast<const float4*>(p + 4);
  v[o + 0] = (_Float16)(a.x * sc); v[o + 1] = (_Float16)(a.y * sc);
  v[o + 2] = (_Float16)(a.z * sc); v[o + 3] = (_Float16)(a.w * sc);
  v[o + 4] = (_Float16)(b.x * sc); v[o + 5] = (_Float16)(b.y * sc);
  v[o + 6] = (_Float16)(b.z * sc); v[o + 7] = (_Float16)(b.w * sc);
}

__device__ __forceinline__ v2h pkmax(v2h a, v2h b) {
#if __has_builtin(__builtin_elementwise_max)
  return __builtin_elementwise_max(a, b);   // v_pk_max_f16
#else
  v2h r; r[0] = (a[0] > b[0]) ? a[0] : b[0]; r[1] = (a[1] > b[1]) ? a[1] : b[1];
  return r;
#endif
}

// Packed (2 rows at once) f16 butterfly max over 16-lane halves.
__device__ __forceinline__ v2h redmax16_pk(v2h x) {
#pragma unroll
  for (int m = 1; m <= 8; m <<= 1) {
    int xi = __builtin_bit_cast(int, x);
    int yi = __shfl_xor(xi, m);
    x = pkmax(x, __builtin_bit_cast(v2h, yi));
  }
  return x;
}

// Cooperative f32->f16 staging of one K/V chunk into a given LDS buffer.
__device__ __forceinline__ void coop_load_chunk(const float* __restrict__ head,
                                                int j0,
                                                _Float16 (*K)[kStr],
                                                _Float16 (*VT)[kStr],
                                                int tid) {
  const int krow = tid >> 1;            // 0..63 : K row
  const int kcol = (tid & 1) * 32;      // 0/32  : K col half
  const float* ks = head + (size_t)(j0 + krow) * kRow + kKD + kcol;
  cvt16_store(ks,      &K[krow][kcol]);
  cvt16_store(ks + 16, &K[krow][kcol + 16]);

  const int vrp = tid & 31;             // 0..31 : V row pair
  const int vdh = (tid >> 5) * 16;      // 0/16/32/48 : V d slice
  const float* v0p = head + (size_t)(j0 + 2 * vrp) * kRow + 2 * kKD + vdh;
  const float* v1p = v0p + kRow;
#pragma unroll
  for (int i = 0; i < 16; i += 4) {
    const float4 x0 = *reinterpret_cast<const float4*>(v0p + i);
    const float4 x1 = *reinterpret_cast<const float4*>(v1p + i);
    v2h t0; t0[0] = (_Float16)x0.x; t0[1] = (_Float16)x1.x;
    v2h t1; t1[0] = (_Float16)x0.y; t1[1] = (_Float16)x1.y;
    v2h t2; t2[0] = (_Float16)x0.z; t2[1] = (_Float16)x1.z;
    v2h t3; t3[0] = (_Float16)x0.w; t3[1] = (_Float16)x1.w;
    *reinterpret_cast<v2h*>(&VT[vdh + i + 0][2 * vrp]) = t0;
    *reinterpret_cast<v2h*>(&VT[vdh + i + 1][2 * vrp]) = t1;
    *reinterpret_cast<v2h*>(&VT[vdh + i + 2][2 * vrp]) = t2;
    *reinterpret_cast<v2h*>(&VT[vdh + i + 3][2 * vrp]) = t3;
  }
}

__global__ __launch_bounds__(kWaves * 32)
void fa_fwd_f16wmma(const float* __restrict__ qkv,
                    const float* __restrict__ sp,
                    float* __restrict__ out) {
  // Double-buffered shared K/V chunk; per-wave P tiles.
  __shared__ _Float16 ldsK [2][kChunk][kStr];    // K rows  [m][d]  (f16)
  __shared__ _Float16 ldsVT[2][kKD][kStr];       // V^T     [d][m]  (f16)
  __shared__ _Float16 ldsP [kWaves][16][kStr];   // P tiles [row][m] (f16)

  const int tid  = threadIdx.x;
  const int lane = tid & 31;
  const int wave = tid >> 5;

  const int bid = blockIdx.x;          // 0 .. 4095
  const int qg  = bid & 15;            // 16 Q-groups of 64 rows
  const int h   = (bid >> 4) & 15;
  const int s   = bid >> 8;
  const int q0  = qg * 64 + wave * 16; // this wave's 16 Q rows

  const int  lr   = lane & 15;
  const bool hi   = lane >= 16;
  const int  ksel = hi ? 8 : 0;        // B-fragment K offset for upper lanes
  const int  kb2  = hi ? 8  : 0;       // A-fragment K offset for upper lanes

  // fold sdpa scale (1/8) and log2(e) into Q scale; softmax done in base-2
  const float qs = (fabsf(sp[0]) + 1.0f) * 0.125f * 1.44269504088896340736f;

  const float* __restrict__ head = qkv + (size_t)s * kL * kRow + (size_t)h * kQKV;

  // ---- Q A-fragments (16x32 f16 each), ISA 16-bit A layout ----------------
  const float* qp = head + (size_t)(q0 + lr) * kRow + kb2;
  v16h aQ0, aQ1;
  fill8_scaled(qp +  0, qs, aQ0, 0);
  fill8_scaled(qp + 16, qs, aQ0, 8);
  fill8_scaled(qp + 32, qs, aQ1, 0);
  fill8_scaled(qp + 48, qs, aQ1, 8);

  // ones B-fragment for WMMA row-sum (P . 1 -> row sums in C layout)
  v16h bOnes;
#pragma unroll
  for (int i = 0; i < 16; ++i) bOnes[i] = (_Float16)1.0f;

  // ---- flash-attention state ----------------------------------------------
  float m_i[8], l_i[8];
#pragma unroll
  for (int r = 0; r < 8; ++r) { m_i[r] = -1.0e30f; l_i[r] = 0.0f; }
  v8f acc[4];
#pragma unroll
  for (int t = 0; t < 4; ++t) acc[t] = {};

  // preload chunk 0 into buffer 0
  coop_load_chunk(head, 0, ldsK[0], ldsVT[0], tid);
  __syncthreads();

  for (int it = 0; it < kIters; ++it) {
    const int cur = it & 1;

    // ---- software pipeline: stage chunk it+1 into the other buffer --------
    if (it + 1 < kIters)
      coop_load_chunk(head, (it + 1) * kChunk, ldsK[cur ^ 1], ldsVT[cur ^ 1], tid);

    // ================= S = Q.K^T (4 col groups x K-dim 64) =================
    v8f sf[4];
#pragma unroll
    for (int c = 0; c < 4; ++c) {
      const v16h bKlo = ld_frag16(&ldsK[cur][c * 16 + lr][ksel]);
      const v16h bKhi = ld_frag16(&ldsK[cur][c * 16 + lr][32 + ksel]);
      v8f z = {};
      v8f t = __builtin_amdgcn_wmma_f32_16x16x32_f16(false, aQ0, false, bKlo,
                                                     (short)0, z, false, false);
      sf[c] = __builtin_amdgcn_wmma_f32_16x16x32_f16(false, aQ1, false, bKhi,
                                                     (short)0, t, false, false);
    }

    // ---- tile row-max: packed f16 butterfly (2 rows per shuffle) ----------
    float tmax[8];
#pragma unroll
    for (int rp = 0; rp < 4; ++rp) {
      const int r0 = 2 * rp, r1 = 2 * rp + 1;
      v2h mp;
      mp[0] = (_Float16)fmaxf(fmaxf(sf[0][r0], sf[1][r0]),
                              fmaxf(sf[2][r0], sf[3][r0]));
      mp[1] = (_Float16)fmaxf(fmaxf(sf[0][r1], sf[1][r1]),
                              fmaxf(sf[2][r1], sf[3][r1]));
      const v2h mr = redmax16_pk(mp);
      tmax[r0] = (float)mr[0];
      tmax[r1] = (float)mr[1];
    }

    // ---- online softmax update (base-2) -----------------------------------
#pragma unroll
    for (int r = 0; r < 8; ++r) {
      const float mn    = fmaxf(m_i[r], tmax[r]);
      const float alpha = exp2f(m_i[r] - mn);
      m_i[r] = mn;
      l_i[r] *= alpha;                  // row-sum added after WMMA row-sum below
      acc[0][r] *= alpha; acc[1][r] *= alpha;
      acc[2][r] *= alpha; acc[3][r] *= alpha;
      const int R = hi ? r + 8 : r;     // C layout: lane half -> row half
#pragma unroll
      for (int c = 0; c < 4; ++c)
        ldsP[wave][R][c * 16 + lr] = (_Float16)exp2f(sf[c][r] - mn);
    }

    WAVE_LDS_SYNC();  // wave-private P tile ready

    // ---- P back as A-fragments --------------------------------------------
    const v16h aP0 = ld_fragA(&ldsP[wave][lr][0],  kb2);  // m 0..31
    const v16h aP1 = ld_fragA(&ldsP[wave][lr][32], kb2);  // m 32..63

    // row sums of P via matrix pipe: l += (P . 1)
    {
      v8f z = {};
      v8f sm = __builtin_amdgcn_wmma_f32_16x16x32_f16(false, aP0, false, bOnes,
                                                      (short)0, z, false, false);
      sm     = __builtin_amdgcn_wmma_f32_16x16x32_f16(false, aP1, false, bOnes,
                                                      (short)0, sm, false, false);
#pragma unroll
      for (int r = 0; r < 8; ++r) l_i[r] += sm[r];
    }

    // ---- O += P.V : 4 d-slices x 2 m-halves -------------------------------
#pragma unroll
    for (int t = 0; t < 4; ++t) {
      const v16h bV0 = ld_frag16(&ldsVT[cur][t * 16 + lr][ksel]);       // m 0..31
      const v16h bV1 = ld_frag16(&ldsVT[cur][t * 16 + lr][32 + ksel]);  // m 32..63
      acc[t] = __builtin_amdgcn_wmma_f32_16x16x32_f16(false, aP0, false, bV0,
                                                      (short)0, acc[t], false, false);
      acc[t] = __builtin_amdgcn_wmma_f32_16x16x32_f16(false, aP1, false, bV1,
                                                      (short)0, acc[t], false, false);
    }

    // single barrier per iteration: buffer cur free for it+2's staging,
    // buffer cur^1 fully staged for it+1's compute
    __syncthreads();
  }

  // ---- normalize and write out ---------------------------------------------
  float* __restrict__ op = out + (size_t)(s * kL + q0) * kORow + (size_t)h * kKD + lr;
#pragma unroll
  for (int r = 0; r < 8; ++r) {
    const int   R   = hi ? r + 8 : r;
    const float inv = 1.0f / l_i[r];
    float* orow = op + (size_t)R * kORow;
    orow[0]  = acc[0][r] * inv;
    orow[16] = acc[1][r] * inv;
    orow[32] = acc[2][r] * inv;
    orow[48] = acc[3][r] * inv;
  }
}

extern "C" void kernel_launch(void* const* d_in, const int* in_sizes, int n_in,
                              void* d_out, int out_size, void* d_ws, size_t ws_size,
                              hipStream_t stream) {
  (void)in_sizes; (void)n_in; (void)out_size; (void)d_ws; (void)ws_size;
  const float* qkv = (const float*)d_in[0];
  const float* s   = (const float*)d_in[1];
  float*       out = (float*)d_out;

  const int blocks = kNSeq * kH * (kL / kChunk) / 1;   // 16*16*16 = 4096
  fa_fwd_f16wmma<<<blocks, kWaves * 32, 0, stream>>>(qkv, s, out);
}
